// stride_PAM_Module_36301063586281
// MI455X (gfx1250) — matrix operand order repaired and checked by reading, not compiled, not run on hardware
//
#include <hip/hip_runtime.h>

// ---------------------------------------------------------------------------
// CDNA5 (gfx1250) PAM attention module, all GEMMs via V_WMMA_F32_16X16X32_BF16
// K-contiguous operand layouts -> b128 fragment loads; register-blocked tiles;
// async global->LDS double-buffering of q in the flash-attention kernel.
// ---------------------------------------------------------------------------

typedef __bf16 bf16;
typedef __attribute__((ext_vector_type(16))) __bf16 v16bf;
typedef __attribute__((ext_vector_type(8)))  __bf16 v8bf;
typedef __attribute__((ext_vector_type(8)))  float  v8f;
typedef __attribute__((ext_vector_type(4)))  int    v4i;

#define BATCH 8
#define CIN 512
#define HH 64
#define WW 64
#define NPIX 4096   // 64*64
#define KD 64
#define VD 256
#define NK 1024     // 32*32
#define EPSBN 1e-5f

#ifndef __has_builtin
#define __has_builtin(x) 0
#endif
#if defined(__AMDGCN__) && __has_builtin(__builtin_amdgcn_global_load_async_to_lds_b128)
#define HAVE_ASYNC_LDS 1
#else
#define HAVE_ASYNC_LDS 0
#endif

__device__ __forceinline__ v8f wmma_bf16(v16bf a, v16bf b, v8f c) {
  // D = A(16x32) * B(32x16) + C, f32 accumulate
  return __builtin_amdgcn_wmma_f32_16x16x32_bf16(false, a, false, b, (short)0, c,
                                                 false, false);
}

// A-fragment from a K-contiguous row (16-bit A 16x32):
// lane group g: elements 0..15 = K g*16..g*16+15 (matches B fragment order).
__device__ __forceinline__ v16bf load_a_row(const bf16* row, int g) {
  v8bf lo = *(const v8bf*)(row + g * 16);
  v8bf hi = *(const v8bf*)(row + g * 16 + 8);
  return __builtin_shufflevector(lo, hi, 0, 1, 2, 3, 4, 5, 6, 7, 8, 9, 10, 11,
                                 12, 13, 14, 15);
}

// B-fragment from a K-contiguous column (16-bit B 32x16):
// lane group g: elements 0..15 = K g*16..g*16+15 (one 32-byte run).
__device__ __forceinline__ v16bf load_b_col(const bf16* col, int g) {
  return *(const v16bf*)(col + g * 16);
}

__device__ __forceinline__ v16bf load_b_col_pred(const bf16* col, int g,
                                                 bool ok) {
  v16bf z = {};
  if (ok) z = *(const v16bf*)(col + g * 16);  // EXEC-masked load
  return z;
}

// 16B global -> LDS copy; async (ASYNCcnt-tracked, no VGPR round trip) when
// the gfx1250 builtin is available.
__device__ __forceinline__ void copy16_g2l(const bf16* g, bf16* l) {
#if HAVE_ASYNC_LDS
  __builtin_amdgcn_global_load_async_to_lds_b128(
      (__attribute__((address_space(1))) v4i*)(void*)g,
      (__attribute__((address_space(3))) v4i*)(void*)l, 0, 0);
#else
  *(v8bf*)l = *(const v8bf*)g;
#endif
}

__device__ __forceinline__ void wait_async_le(int pending8) {
#if HAVE_ASYNC_LDS
  if (pending8)
    asm volatile("s_wait_asynccnt 8" ::: "memory");
  else
    asm volatile("s_wait_asynccnt 0" ::: "memory");
#else
  (void)pending8;
#endif
}

// ---------------------------------------------------------------------------
// x (NCHW f32) -> xT (NHWC bf16), LDS-tiled transpose, one HBM pass.
// ---------------------------------------------------------------------------
__global__ __launch_bounds__(256) void xpose_kernel(const float* __restrict__ src,
                                                    bf16* __restrict__ dst) {
  __shared__ bf16 tile[32][33];
  int pt = blockIdx.x * 32, ct = blockIdx.y * 32, b = blockIdx.z;
  int tx = threadIdx.x, ty = threadIdx.y;
#pragma unroll
  for (int j = ty; j < 32; j += 8)
    tile[j][tx] = (bf16)src[((size_t)b * CIN + ct + j) * NPIX + pt + tx];
  __syncthreads();
#pragma unroll
  for (int j = ty; j < 32; j += 8)
    dst[((size_t)b * NPIX + pt + j) * CIN + ct + tx] = tile[tx][j];
}

__global__ void cvt_kernel(const float* __restrict__ src, bf16* __restrict__ dst,
                           int n) {
  int i = blockIdx.x * blockDim.x + threadIdx.x;
  if (i < n) dst[i] = (bf16)src[i];
}

// w_k [kd][c][3][3] -> wkr [tap][kd][c]
__global__ void repack_wk_kernel(const float* __restrict__ src,
                                 bf16* __restrict__ dst) {
  int o = blockIdx.x * 256 + threadIdx.x;
  if (o >= 9 * KD * CIN) return;
  int t = o / (KD * CIN), m = (o / CIN) % KD, c = o % CIN;
  dst[o] = (bf16)src[((size_t)m * CIN + c) * 9 + t];
}

// w_d [ci][co][3][3] -> wdr [tap][co][ci]
__global__ void repack_wd_kernel(const float* __restrict__ src,
                                 bf16* __restrict__ dst) {
  int o = blockIdx.x * 256 + threadIdx.x;
  if (o >= 9 * CIN * VD) return;
  int t = o / (CIN * VD), co = (o / VD) % CIN, ci = o % VD;
  dst[o] = (bf16)src[((size_t)ci * CIN + co) * 9 + t];
}

// ---------------------------------------------------------------------------
// q = BN(Wq @ x + bq) -> qT[b][4096][64]. Wave computes 16(m) x 64(n):
// per K-chunk 1 A-frag + 4 B-frags -> 4 WMMAs (A reuse x4).
// ---------------------------------------------------------------------------
__global__ __launch_bounds__(256) void q_kernel(
    const bf16* __restrict__ xT, const bf16* __restrict__ wqb,
    const float* __restrict__ bq, const float* __restrict__ bs,
    const float* __restrict__ bb, const float* __restrict__ bm,
    const float* __restrict__ bv, bf16* __restrict__ qT) {
  int tid = threadIdx.x, wave = tid >> 5, lane = tid & 31;
  int g = lane >> 4, ln = lane & 15;
  int wid = blockIdx.x * 8 + wave;                    // 2048 waves
  int b = wid >> 8, mt = (wid >> 6) & 3, nt = wid & 63;
  int m0 = mt * 16, n0 = nt * 64;
  const bf16* wrow = wqb + (size_t)(m0 + ln) * CIN;
  const bf16* xr[4];
#pragma unroll
  for (int j = 0; j < 4; ++j)
    xr[j] = xT + (size_t)(b * NPIX + n0 + j * 16 + ln) * CIN;
  v8f acc[4] = {};
#pragma unroll 1
  for (int c = 0; c < CIN; c += 32) {
    v16bf a = load_a_row(wrow + c, g);
#pragma unroll
    for (int j = 0; j < 4; ++j) {
      __builtin_prefetch(xr[j] + c + 64);
      acc[j] = wmma_bf16(a, load_b_col(xr[j] + c, g), acc[j]);
    }
  }
  v8bf pk[4];
#pragma unroll
  for (int r = 0; r < 8; ++r) {
    int m = m0 + r + 8 * g;
    float al = bs[m] * rsqrtf(bv[m] + EPSBN);
    float be = bq[m] * al + bb[m] - bm[m] * al;
#pragma unroll
    for (int j = 0; j < 4; ++j) pk[j][r] = (bf16)(acc[j][r] * al + be);
  }
#pragma unroll
  for (int j = 0; j < 4; ++j)
    *(v8bf*)(qT + (size_t)(b * NPIX + n0 + j * 16 + ln) * KD + m0 + 8 * g) =
        pk[j];
}

// ---------------------------------------------------------------------------
// k = BN(conv3x3_s2(x) + bk) -> ktb[b][1024][64]. Wave computes 16 x 64.
// ---------------------------------------------------------------------------
__global__ __launch_bounds__(256) void k_kernel(
    const bf16* __restrict__ xT, const bf16* __restrict__ wkr,
    const float* __restrict__ bk, const float* __restrict__ bs,
    const float* __restrict__ bb, const float* __restrict__ bm,
    const float* __restrict__ bv, bf16* __restrict__ ktb) {
  int tid = threadIdx.x, wave = tid >> 5, lane = tid & 31;
  int g = lane >> 4, ln = lane & 15;
  int wid = blockIdx.x * 8 + wave;                    // 512 waves
  int b = wid >> 6, mt = (wid >> 4) & 3, nt = wid & 15;
  int m0 = mt * 16, n0 = nt * 64;
  int msp[4], mh[4], mw[4];
#pragma unroll
  for (int j = 0; j < 4; ++j) {
    msp[j] = n0 + j * 16 + ln;
    mh[j] = msp[j] >> 5;
    mw[j] = msp[j] & 31;
  }
  v8f acc[4] = {};
#pragma unroll 1
  for (int kh = 0; kh < 3; ++kh) {
#pragma unroll 1
    for (int kw = 0; kw < 3; ++kw) {
      const bf16* wrow = wkr + (size_t)((kh * 3 + kw) * KD + m0 + ln) * CIN;
      const bf16* xr[4];
      bool ok[4];
#pragma unroll
      for (int j = 0; j < 4; ++j) {
        int ih = 2 * mh[j] - 1 + kh, iw = 2 * mw[j] - 1 + kw;
        ok[j] = ((unsigned)ih < 64u) && ((unsigned)iw < 64u);
        int sp = ok[j] ? (ih * WW + iw) : 0;
        xr[j] = xT + (size_t)(b * NPIX + sp) * CIN;
      }
#pragma unroll 1
      for (int c = 0; c < CIN; c += 32) {
        v16bf a = load_a_row(wrow + c, g);
#pragma unroll
        for (int j = 0; j < 4; ++j)
          acc[j] = wmma_bf16(a, load_b_col_pred(xr[j] + c, g, ok[j]), acc[j]);
      }
    }
  }
  v8bf pk[4];
#pragma unroll
  for (int r = 0; r < 8; ++r) {
    int m = m0 + r + 8 * g;
    float al = bs[m] * rsqrtf(bv[m] + EPSBN);
    float be = bk[m] * al + bb[m] - bm[m] * al;
#pragma unroll
    for (int j = 0; j < 4; ++j) pk[j][r] = (bf16)(acc[j][r] * al + be);
  }
#pragma unroll
  for (int j = 0; j < 4; ++j)
    *(v8bf*)(ktb + (size_t)(b * NK + msp[j]) * KD + m0 + 8 * g) = pk[j];
}

// ---------------------------------------------------------------------------
// v = Wv @ x + bv -> vb[b][256][4096]. Wave computes 32(m) x 32(n):
// per K-chunk 2 A + 2 B -> 4 WMMAs.
// ---------------------------------------------------------------------------
__global__ __launch_bounds__(256) void v_kernel(const bf16* __restrict__ xT,
                                                const bf16* __restrict__ wvb,
                                                const float* __restrict__ bvv,
                                                bf16* __restrict__ vb) {
  int tid = threadIdx.x, wave = tid >> 5, lane = tid & 31;
  int g = lane >> 4, ln = lane & 15;
  int wid = blockIdx.x * 8 + wave;                    // 8192 waves
  int b = wid >> 10, mt = (wid >> 7) & 7, nt = wid & 127;
  int m0 = mt * 32, n0 = nt * 32;
  const bf16* wr0 = wvb + (size_t)(m0 + ln) * CIN;
  const bf16* wr1 = wvb + (size_t)(m0 + 16 + ln) * CIN;
  const bf16* x0 = xT + (size_t)(b * NPIX + n0 + ln) * CIN;
  const bf16* x1 = xT + (size_t)(b * NPIX + n0 + 16 + ln) * CIN;
  v8f a00 = {}, a01 = {}, a10 = {}, a11 = {};
#pragma unroll 1
  for (int c = 0; c < CIN; c += 32) {
    __builtin_prefetch(x0 + c + 64);
    __builtin_prefetch(x1 + c + 64);
    v16bf fa0 = load_a_row(wr0 + c, g);
    v16bf fa1 = load_a_row(wr1 + c, g);
    v16bf fb0 = load_b_col(x0 + c, g);
    v16bf fb1 = load_b_col(x1 + c, g);
    a00 = wmma_bf16(fa0, fb0, a00);
    a01 = wmma_bf16(fa0, fb1, a01);
    a10 = wmma_bf16(fa1, fb0, a10);
    a11 = wmma_bf16(fa1, fb1, a11);
  }
#pragma unroll
  for (int r = 0; r < 8; ++r) {
    int ma = m0 + r + 8 * g, mb = m0 + 16 + r + 8 * g;
    float ba = bvv[ma], bbv = bvv[mb];
    vb[(size_t)(b * VD + ma) * NPIX + n0 + ln] = (bf16)(a00[r] + ba);
    vb[(size_t)(b * VD + ma) * NPIX + n0 + 16 + ln] = (bf16)(a01[r] + ba);
    vb[(size_t)(b * VD + mb) * NPIX + n0 + ln] = (bf16)(a10[r] + bbv);
    vb[(size_t)(b * VD + mb) * NPIX + n0 + 16 + ln] = (bf16)(a11[r] + bbv);
  }
}

// ---------------------------------------------------------------------------
// Flash attention with async-LDS double-buffered q slices.
// WG = (b, 16-row m-tile); wave w owns n-slice [w*32,+32) of each 256-wide
// chunk and vd-slice [w*32,+32) of the output. q slice (32px x 64d = 4KB)
// is staged per wave into LDS with GLOBAL_LOAD_ASYNC_TO_LDS_B128, double
// buffered across iterations (s_wait_asynccnt 8 = previous chunk complete).
// ---------------------------------------------------------------------------
__global__ __launch_bounds__(256) void attn_kernel(
    const bf16* __restrict__ qT, const bf16* __restrict__ ktb,
    const bf16* __restrict__ vb, bf16* __restrict__ attp) {
  __shared__ float sS[16 * 256];
  __shared__ __attribute__((aligned(32))) bf16 sP[16 * 256];
  __shared__ __attribute__((aligned(16))) bf16 sQ[2 * 8 * 2048];  // 64KB
  __shared__ float sMax[16], sSum[16], sScale[16];
  int b = blockIdx.y, m0 = blockIdx.x * 16;
  int tid = threadIdx.x, wave = tid >> 5, lane = tid & 31;
  int g = lane >> 4, ln = lane & 15;

  const bf16* ktrow = ktb + (size_t)(b * NK + m0 + ln) * KD;
  v16bf a0 = load_a_row(ktrow, g);       // d = 0..31
  v16bf a1 = load_a_row(ktrow + 32, g);  // d = 32..63

  v8f o0 = {}, o1 = {};
  if (tid < 16) { sMax[tid] = -1e30f; sSum[tid] = 0.0f; }

  const bf16* qg = qT + (size_t)(b * NPIX + wave * 32) * KD;  // wave's column
  // stage iteration 0 (8 x 16B per lane, 4KB per wave)
#pragma unroll
  for (int i = 0; i < 8; ++i) {
    int off = (i * 32 + lane) * 8;
    copy16_g2l(qg + off, sQ + (size_t)wave * 2048 + off);
  }

#pragma unroll 1
  for (int it = 0; it < 16; ++it) {
    int cur = it & 1;
    if (it < 15) {  // stage next iteration's q slice into the other buffer
      const bf16* gsrc = qg + (size_t)(it + 1) * 256 * KD;
      bf16* ldst = sQ + (size_t)((cur ^ 1) * 8 + wave) * 2048;
#pragma unroll
      for (int i = 0; i < 8; ++i) {
        int off = (i * 32 + lane) * 8;
        copy16_g2l(gsrc + off, ldst + off);
      }
      wait_async_le(1);  // <=8 outstanding: current buffer complete
    } else {
      wait_async_le(0);
    }

    const bf16* qw = sQ + (size_t)(cur * 8 + wave) * 2048;
    v16bf b00 = load_b_col(qw + ln * 64, g);             // n-sub 0, d 0..31
    v16bf b10 = load_b_col(qw + ln * 64 + 32, g);        // n-sub 0, d 32..63
    v16bf b01 = load_b_col(qw + (16 + ln) * 64, g);      // n-sub 1
    v16bf b11 = load_b_col(qw + (16 + ln) * 64 + 32, g);
    v8f s0 = {}, s1 = {};
    s0 = wmma_bf16(a0, b00, s0);
    s0 = wmma_bf16(a1, b10, s0);
    s1 = wmma_bf16(a0, b01, s1);
    s1 = wmma_bf16(a1, b11, s1);

    __syncthreads();  // previous iteration done reading sS/sP
#pragma unroll
    for (int r = 0; r < 8; ++r) {
      int m = r + 8 * g;
      sS[m * 256 + wave * 32 + ln] = s0[r];
      sS[m * 256 + wave * 32 + 16 + ln] = s1[r];
    }
    __syncthreads();

    if (tid < 16) {  // online softmax row stats
      int m = tid;
      float mx = sMax[m], cm = mx;
      for (int c2 = 0; c2 < 256; ++c2) cm = fmaxf(cm, sS[m * 256 + c2]);
      float sc = __expf(mx - cm);
      float sm = sSum[m] * sc;
      for (int c2 = 0; c2 < 256; ++c2) sm += __expf(sS[m * 256 + c2] - cm);
      sMax[m] = cm;
      sScale[m] = sc;
      sSum[m] = sm;
    }
    __syncthreads();

    // P = exp(S - rowmax) (coalesced), rescale O accumulators
#pragma unroll 1
    for (int m = 0; m < 16; ++m)
      sP[m * 256 + tid] = (bf16)__expf(sS[m * 256 + tid] - sMax[m]);
#pragma unroll
    for (int r = 0; r < 8; ++r) {
      float sc = sScale[r + 8 * g];
      o0[r] *= sc;
      o1[r] *= sc;
    }
    __syncthreads();

    // O(16 x 32 vd-slice) += P(16 x 256) @ V^T(256 x 32)
#pragma unroll 1
    for (int kk = 0; kk < 8; ++kk) {
      v16bf ap = load_a_row((const bf16*)sP + ln * 256 + kk * 32, g);
      const bf16* vc0 =
          vb + (size_t)(b * VD + wave * 32 + ln) * NPIX + it * 256 + kk * 32;
      const bf16* vc1 =
          vb + (size_t)(b * VD + wave * 32 + 16 + ln) * NPIX + it * 256 + kk * 32;
      o0 = wmma_bf16(ap, load_b_col(vc0, g), o0);
      o1 = wmma_bf16(ap, load_b_col(vc1, g), o1);
    }
  }
  __syncthreads();

#pragma unroll
  for (int r = 0; r < 8; ++r) {
    int m = r + 8 * g;
    float inv = 1.0f / sSum[m];
    attp[(size_t)(b * NK + m0 + m) * VD + wave * 32 + ln] = (bf16)(o0[r] * inv);
    attp[(size_t)(b * NK + m0 + m) * VD + wave * 32 + 16 + ln] =
        (bf16)(o1[r] * inv);
  }
}

// ---------------------------------------------------------------------------
// Transpose conv 3x3 stride2 (256->512) + BN + ReLU + residual, fused.
// Parity decomposition; wave computes 16(co) x 64(px): per tap per K-chunk
// 1 A + 4 B -> 4 WMMAs.
// ---------------------------------------------------------------------------
__global__ __launch_bounds__(256) void deconv_kernel(
    const bf16* __restrict__ attp, const bf16* __restrict__ wdr,
    const float* __restrict__ bs, const float* __restrict__ bb,
    const float* __restrict__ bm, const float* __restrict__ bv,
    const float* __restrict__ x, float* __restrict__ out) {
  int tid = threadIdx.x, wave = tid >> 5, lane = tid & 31;
  int g = lane >> 4, ln = lane & 15;
  int wid = blockIdx.x * 8 + wave;                    // 16384 waves
  int b = wid >> 11;
  int par = (wid >> 9) & 3;
  int ct = (wid >> 4) & 31;
  int pt = wid & 15;
  int ph = par >> 1, pw = par & 1;
  int m0 = ct * 16;        // output channel base
  int ihp[4], iwp[4], oh[4], ow[4];
#pragma unroll
  for (int j = 0; j < 4; ++j) {
    int p = pt * 64 + j * 16 + ln;
    ihp[j] = p >> 5;
    iwp[j] = p & 31;
    oh[j] = 2 * ihp[j] + ph;
    ow[j] = 2 * iwp[j] + pw;
  }

  int kh_a[2], nh;
  if (ph == 0) { nh = 1; kh_a[0] = 1; }
  else { nh = 2; kh_a[0] = 0; kh_a[1] = 2; }
  int kw_a[2], nw;
  if (pw == 0) { nw = 1; kw_a[0] = 1; }
  else { nw = 2; kw_a[0] = 0; kw_a[1] = 2; }

  v8f acc[4] = {};
#pragma unroll 1
  for (int ta = 0; ta < nh; ++ta) {
#pragma unroll 1
    for (int tb = 0; tb < nw; ++tb) {
      int kh = kh_a[ta], kw = kw_a[tb];
      int dh = (ph == 1 && kh == 0) ? 1 : 0;  // ih = ihp + dh
      int dw = (pw == 1 && kw == 0) ? 1 : 0;  // iw = iwp + dw
      const bf16* wrow = wdr + (size_t)((kh * 3 + kw) * CIN + m0 + ln) * VD;
      const bf16* br[4];
      bool ok[4];
#pragma unroll
      for (int j = 0; j < 4; ++j) {
        int ih = ihp[j] + dh, iw = iwp[j] + dw;
        ok[j] = (ih < 32) && (iw < 32);
        int sp = ok[j] ? (ih * 32 + iw) : 0;
        br[j] = attp + (size_t)(b * NK + sp) * VD;
      }
#pragma unroll 1
      for (int c = 0; c < VD; c += 32) {
        v16bf a = load_a_row(wrow + c, g);
#pragma unroll
        for (int j = 0; j < 4; ++j)
          acc[j] = wmma_bf16(a, load_b_col_pred(br[j] + c, g, ok[j]), acc[j]);
      }
    }
  }
#pragma unroll
  for (int r = 0; r < 8; ++r) {
    int co = m0 + r + 8 * g;
    float al = bs[co] * rsqrtf(bv[co] + EPSBN);
    float be = bb[co] - bm[co] * al;
#pragma unroll
    for (int j = 0; j < 4; ++j) {
      float y = fmaxf(acc[j][r] * al + be, 0.0f);
      size_t oi = (((size_t)b * CIN + co) * HH + oh[j]) * WW + ow[j];
      out[oi] = y + x[oi];
    }
  }
}

// ---------------------------------------------------------------------------
// host launcher
// ---------------------------------------------------------------------------
extern "C" void kernel_launch(void* const* d_in, const int* in_sizes, int n_in,
                              void* d_out, int out_size, void* d_ws,
                              size_t ws_size, hipStream_t stream) {
  (void)in_sizes; (void)n_in; (void)out_size; (void)ws_size;
  const float* x    = (const float*)d_in[0];
  const float* w_q  = (const float*)d_in[1];
  const float* b_q  = (const float*)d_in[2];
  const float* bqs  = (const float*)d_in[3];
  const float* bqb  = (const float*)d_in[4];
  const float* bqm  = (const float*)d_in[5];
  const float* bqv  = (const float*)d_in[6];
  const float* w_k  = (const float*)d_in[7];
  const float* b_k  = (const float*)d_in[8];
  const float* bks  = (const float*)d_in[9];
  const float* bkb  = (const float*)d_in[10];
  const float* bkm  = (const float*)d_in[11];
  const float* bkv  = (const float*)d_in[12];
  const float* w_v  = (const float*)d_in[13];
  const float* b_v  = (const float*)d_in[14];
  const float* w_d  = (const float*)d_in[15];
  const float* bds  = (const float*)d_in[16];
  const float* bdb  = (const float*)d_in[17];
  const float* bdm  = (const float*)d_in[18];
  const float* bdv  = (const float*)d_in[19];
  float* out = (float*)d_out;

  char* ws = (char*)d_ws;
  size_t off = 0;
  const size_t nXT  = (size_t)BATCH * NPIX * CIN;
  const size_t nWQ  = (size_t)KD * CIN;
  const size_t nWK  = (size_t)9 * KD * CIN;
  const size_t nWV  = (size_t)VD * CIN;
  const size_t nWD  = (size_t)9 * CIN * VD;
  const size_t nQ   = (size_t)BATCH * NPIX * KD;
  const size_t nKT  = (size_t)BATCH * NK * KD;
  const size_t nV   = (size_t)BATCH * VD * NPIX;
  const size_t nAT  = (size_t)BATCH * NK * VD;
  bf16* xT  = (bf16*)(ws + off); off += nXT * 2;
  bf16* wqb = (bf16*)(ws + off); off += nWQ * 2;
  bf16* wkr = (bf16*)(ws + off); off += nWK * 2;
  bf16* wvb = (bf16*)(ws + off); off += nWV * 2;
  bf16* wdr = (bf16*)(ws + off); off += nWD * 2;
  bf16* qT  = (bf16*)(ws + off); off += nQ * 2;
  bf16* ktb = (bf16*)(ws + off); off += nKT * 2;
  bf16* vbf = (bf16*)(ws + off); off += nV * 2;
  bf16* atp = (bf16*)(ws + off); off += nAT * 2;

  xpose_kernel<<<dim3(NPIX / 32, CIN / 32, BATCH), dim3(32, 8), 0, stream>>>(x, xT);
  cvt_kernel<<<(unsigned)((nWQ + 255) / 256), 256, 0, stream>>>(w_q, wqb, (int)nWQ);
  cvt_kernel<<<(unsigned)((nWV + 255) / 256), 256, 0, stream>>>(w_v, wvb, (int)nWV);
  repack_wk_kernel<<<(unsigned)((nWK + 255) / 256), 256, 0, stream>>>(w_k, wkr);
  repack_wd_kernel<<<(unsigned)((nWD + 255) / 256), 256, 0, stream>>>(w_d, wdr);

  q_kernel<<<256, 256, 0, stream>>>(xT, wqb, b_q, bqs, bqb, bqm, bqv, qT);
  k_kernel<<<64, 256, 0, stream>>>(xT, wkr, b_k, bks, bkb, bkm, bkv, ktb);
  v_kernel<<<1024, 256, 0, stream>>>(xT, wvb, b_v, vbf);
  attn_kernel<<<dim3(64, BATCH), 256, 0, stream>>>(qT, ktb, vbf, atp);
  deconv_kernel<<<2048, 256, 0, stream>>>(atp, wdr, bds, bdb, bdm, bdv, x, out);
}
